// Fai_score_14542759264420
// MI455X (gfx1250) — compile-verified, matrix operand order checked by reading, not run on hardware
//
#include <hip/hip_runtime.h>
#include <hip/hip_bf16.h>
#include <math.h>

// Problem constants (from reference)
#define MM 256
#define KK 64
#define NN 2048
#define LT 20
#define LM 10
#define EE 300
#define VV 5053
#define CCH 128
#define HH 64
#define WW 5
#define FF 623

#define VP 5056     // V padded to multiple of 16
#define EP 320      // E padded to multiple of 32

typedef __attribute__((ext_vector_type(16))) _Float16 v16h;
typedef __attribute__((ext_vector_type(8)))  _Float16 v8h;
typedef __attribute__((ext_vector_type(8)))  float    v8f;

// WMMA 16-bit A/B fragment (ISA 7.12.2): lane<16 holds K in {0..7,16..23},
// lane>=16 holds K in {8..15,24..31}. With K contiguous in memory this is two
// aligned 16-byte loads.
__device__ __forceinline__ v16h load_frag(const _Float16* __restrict__ rowK, int lane) {
  const int lo = (lane & 16) ? 8 : 0;
  v8h a = *(const v8h*)(rowK + lo);
  v8h b = *(const v8h*)(rowK + lo + 16);
  return __builtin_shufflevector(a, b, 0, 1, 2, 3, 4, 5, 6, 7, 8, 9, 10, 11, 12, 13, 14, 15);
}
__device__ __forceinline__ v16h load_frag_lds(const _Float16* rowK, int lane) {
  const int lo = (lane & 16) ? 8 : 0;
  v8h a = *(const v8h*)(rowK + lo);
  v8h b = *(const v8h*)(rowK + lo + 16);
  return __builtin_shufflevector(a, b, 0, 1, 2, 3, 4, 5, 6, 7, 8, 9, 10, 11, 12, 13, 14, 15);
}

__device__ __forceinline__ v8f wmma16(v16h a, v16h b, v8f c) {
  return __builtin_amdgcn_wmma_f32_16x16x32_f16(false, a, false, b, (short)0, c, false, false);
}

__device__ __forceinline__ float sigf(float x) { return 1.f / (1.f + expf(-x)); }

// ---------------------------------------------------------------------------
// P0: char_emb (V x E f32) -> A16p (VP x EP f16, zero padded)
// ---------------------------------------------------------------------------
__global__ void prep_A16(const float* __restrict__ ce, _Float16* __restrict__ A16p) {
  int i = blockIdx.x * 256 + threadIdx.x;
  if (i >= VP * EP) return;
  int v = i / EP, e = i % EP;
  A16p[i] = (v < VV && e < EE) ? (_Float16)ce[v * EE + e] : (_Float16)0.f;
}

// P1: conv_w (CC,E,W f32) -> Bt (640 x EP f16), Bt[n][e] = conv_w[n&127][e][n>>7]
__global__ void prep_Bt(const float* __restrict__ cw, _Float16* __restrict__ Bt) {
  int i = blockIdx.x * 256 + threadIdx.x;
  if (i >= 640 * EP) return;
  int n = i / EP, e = i % EP;
  int cc = n & 127, w = n >> 7;
  Bt[i] = (e < EE) ? (_Float16)cw[(cc * EE + e) * WW + w] : (_Float16)0.f;
}

// P2: transpose f32 [Kd][Nd] -> f16 [Nd][Kd]
__global__ void transpose_f16(const float* __restrict__ src, _Float16* __restrict__ dst,
                              int Kd, int Nd) {
  int i = blockIdx.x * 256 + threadIdx.x;
  if (i >= Kd * Nd) return;
  int n = i / Kd, k = i % Kd;
  dst[i] = (_Float16)src[k * Nd + n];
}

// ---------------------------------------------------------------------------
// K1: token projection table P[v][w*128+cc] = sum_e char_emb[v,e]*conv_w[cc,e,w]
// (VP x 640) GEMM, K=EP, fully padded: guard-free b128 fragment loads.
// 316 M-tiles x 40 N-tiles; 8 waves/block, 1 tile/wave (12640 % 8 == 0).
// ---------------------------------------------------------------------------
__global__ void tokproj_gemm(const _Float16* __restrict__ A16p,
                             const _Float16* __restrict__ Bt,
                             float* __restrict__ P /* VP x 640 */) {
  const int wv = threadIdx.x >> 5, lane = threadIdx.x & 31;
  const int g = blockIdx.x * 8 + wv;
  const int tM = g / 40, tN = g % 40;
  const _Float16* Arow = A16p + (size_t)(tM * 16 + (lane & 15)) * EP;
  const _Float16* Brow = Bt + (size_t)(tN * 16 + (lane & 15)) * EP;
  v8f acc = {};
  for (int k0 = 0; k0 < EP; k0 += 32) {
    v16h a = load_frag(Arow + k0, lane);
    v16h b = load_frag(Brow + k0, lane);
    acc = wmma16(a, b, acc);
  }
  const int colD = tN * 16 + (lane & 15);
  const int rbase = tM * 16 + ((lane >> 4) << 3);
  for (int r = 0; r < 8; ++r) P[(size_t)(rbase + r) * 640 + colD] = acc[r];
}

// ---------------------------------------------------------------------------
// K2: conv+relu+maxpool over a token sequence using the P table.
// grid.x = #sequences, block = 128 (one thread per output channel).
// ---------------------------------------------------------------------------
__global__ void conv_pool_kernel(const int* __restrict__ toks, const float* __restrict__ P,
                                 const float* __restrict__ conv_b, _Float16* __restrict__ out,
                                 int L, int npos) {
  __shared__ int tk[32];
  const int b = blockIdx.x, cc = threadIdx.x;
  if (cc < L) tk[cc] = toks[b * L + cc];
  __syncthreads();
  const float bias = conv_b[cc];
  float best = -1e30f;
  for (int p = 0; p < npos; ++p) {
    float acc = bias;
    for (int w = 0; w < WW; ++w) acc += P[(size_t)tk[p + w] * 640 + w * 128 + cc];
    acc = fmaxf(acc, 0.f);
    best = fmaxf(best, acc);
  }
  out[b * 128 + cc] = (_Float16)best;
}

// ---------------------------------------------------------------------------
// K3: LSTM over the K=64 candidate axis. One workgroup = one (instance, 16-row
// M-tile). inst bit0: direction (1=backward), bit1: input (1=mention/q).
// Weights pre-transposed to [n][K] so B fragments are contiguous b128 loads.
// ---------------------------------------------------------------------------
__global__ void lstm_kernel(const int* __restrict__ cand_idx,
                            const _Float16* __restrict__ T16, const _Float16* __restrict__ Q16,
                            const _Float16* __restrict__ WiTF, const _Float16* __restrict__ WhTF,
                            const _Float16* __restrict__ WiTB, const _Float16* __restrict__ WhTB,
                            const float* __restrict__ bF, const float* __restrict__ bB,
                            float* __restrict__ hs) {
  const int inst = blockIdx.x >> 4;
  const int m0 = (blockIdx.x & 15) * 16;
  const bool bwd = (inst & 1) != 0, qin = (inst & 2) != 0;
  const _Float16* WiT = bwd ? WiTB : WiTF;   // [256][128]
  const _Float16* WhT = bwd ? WhTB : WhTF;   // [256][64]
  const float* bias = bwd ? bB : bF;
  float* out = hs + (long)inst * (MM * KK * HH);

  __shared__ float    G[16 * 256];
  __shared__ __align__(16) _Float16 h16[16 * 64];
  __shared__ float    cst[16 * 64];
  __shared__ int      sIdx[16];

  const int tid = threadIdx.x, lane = tid & 31, wv = tid >> 5;
  for (int i = tid; i < 16 * 64; i += 256) { h16[i] = (_Float16)0.f; cst[i] = 0.f; }

  const int arow = lane & 15;
  const int n0 = (wv * 2) * 16 + (lane & 15);
  const int n1 = n0 + 16;

  for (int step = 0; step < 64; ++step) {
    const int k = bwd ? (63 - step) : step;
    if (!qin && tid < 16) sIdx[tid] = cand_idx[(m0 + tid) * KK + k];
    __syncthreads();

    v8f acc0 = {}, acc1 = {};
    const long xbase = qin ? (long)(m0 + arow) * 128 : (long)sIdx[arow] * 128;
    const _Float16* xsrc = (qin ? Q16 : T16) + xbase;
    for (int k0 = 0; k0 < 128; k0 += 32) {           // x @ Wi
      v16h a  = load_frag(xsrc + k0, lane);
      v16h b0 = load_frag(WiT + (size_t)n0 * 128 + k0, lane);
      v16h b1 = load_frag(WiT + (size_t)n1 * 128 + k0, lane);
      acc0 = wmma16(a, b0, acc0);
      acc1 = wmma16(a, b1, acc1);
    }
    for (int k0 = 0; k0 < 64; k0 += 32) {            // h @ Wh
      v16h a  = load_frag_lds(h16 + arow * 64 + k0, lane);
      v16h b0 = load_frag(WhT + (size_t)n0 * 64 + k0, lane);
      v16h b1 = load_frag(WhT + (size_t)n1 * 64 + k0, lane);
      acc0 = wmma16(a, b0, acc0);
      acc1 = wmma16(a, b1, acc1);
    }
    const int rb = (lane >> 4) << 3;
    for (int r = 0; r < 8; ++r) {
      G[(rb + r) * 256 + n0] = acc0[r];
      G[(rb + r) * 256 + n1] = acc1[r];
    }
    __syncthreads();

    for (int e = 0; e < 4; ++e) {                    // elementwise gates
      int lin = tid + 256 * e;
      int r = lin >> 6, h = lin & 63;
      float gi = G[r * 256 + h]        + bias[h];
      float gf = G[r * 256 + 64 + h]   + bias[64 + h];
      float gg = G[r * 256 + 128 + h]  + bias[128 + h];
      float go = G[r * 256 + 192 + h]  + bias[192 + h];
      float cc = sigf(gf) * cst[r * 64 + h] + sigf(gi) * tanhf(gg);
      float hh = sigf(go) * tanhf(cc);
      cst[r * 64 + h] = cc;
      h16[r * 64 + h] = (_Float16)hh;
      out[((long)(m0 + r) * KK + k) * HH + h] = hh;
    }
    __syncthreads();
  }
}

// ---------------------------------------------------------------------------
// K4: BiDAF + scoring + normalizations + scatter. One workgroup per mention m.
// LDS overlay (60KB):
//  [0,16K)   cF (c f32 64x64)
//  [16K,32K) qF -> S -> c2q (phased reuse)
//  [32K,40K) qT16 (q transposed, f16)   [40K,48K) q16
//  [48K,56K) cs16 -> A16
//  [56K,60K) misc scalars/vectors
// ---------------------------------------------------------------------------
__global__ void bidaf_kernel(const int* __restrict__ cand_idx, const float* __restrict__ sfeats,
                             const float* __restrict__ hs,
                             const float* __restrict__ wc, const float* __restrict__ bc,
                             const float* __restrict__ wq, const float* __restrict__ bq,
                             const float* __restrict__ wcq, const float* __restrict__ bcq,
                             const float* __restrict__ wz, const float* __restrict__ bz,
                             const float* __restrict__ wloc, const float* __restrict__ bloc,
                             float* __restrict__ out) {
  __shared__ __align__(16) char smem[61440];
  float*    cF   = (float*)(smem);
  float*    qF   = (float*)(smem + 16384);   // later: S, then c2q
  float*    S    = qF;
  float*    c2q  = qF;
  _Float16* qT16 = (_Float16*)(smem + 32768); // q transposed: qT16[h][j] = q[j][h]
  _Float16* q16  = (_Float16*)(smem + 40960);
  _Float16* cs16 = (_Float16*)(smem + 49152); // later: A16
  _Float16* A16  = cs16;
  float*    misc = (float*)(smem + 57344);
  float* cw = misc;           float* qw = misc + 64;
  float* smaxA = misc + 128;  float* b2a = misc + 192;
  float* q2c = misc + 256;    float* p1a = misc + 320;
  float* tv = misc + 384;     float* zA = misc + 448;
  float* softA = misc + 512;  float* uniA = misc + 576;
  float* softsA = misc + 640; float* part = misc + 704;   // 256 floats

  const int m = blockIdx.x, tid = threadIdx.x, lane = tid & 31, wv = tid >> 5;
  const long HS = (long)MM * KK * HH;

  // phase 1: assemble c = tf+tb, q = qf+qb
  for (int i = tid; i < 4096; i += 256) {
    float cv = hs[0 * HS + (long)m * 4096 + i] + hs[1 * HS + (long)m * 4096 + i];
    float qv = hs[2 * HS + (long)m * 4096 + i] + hs[3 * HS + (long)m * 4096 + i];
    cF[i] = cv;
    qF[i] = qv;
    q16[i] = (_Float16)qv;
    qT16[(i & 63) * 64 + (i >> 6)] = (_Float16)qv;
  }
  __syncthreads();
  // phase 1b: cs16 = c * wcq (f16), cw/qw row dots
  for (int i = tid; i < 4096; i += 256) cs16[i] = (_Float16)(cF[i] * wcq[i & 63]);
  if (tid < 64) {
    float s1 = bc[0], s2 = bq[0];
    for (int h = 0; h < 64; ++h) { s1 += cF[tid * 64 + h] * wc[h]; s2 += qF[tid * 64 + h] * wq[h]; }
    cw[tid] = s1; qw[tid] = s2;
  }
  __syncthreads();

  // phase 2: S = cs @ q^T + cw_i + qw_j + bcq (WMMA, 16 tiles, 2 per wave)
  const float bcqv = bcq[0];
  for (int t = 2 * wv; t < 2 * wv + 2; ++t) {
    const int ti = t >> 2, tj = t & 3;
    v8f acc = {};
    for (int k0 = 0; k0 < 64; k0 += 32) {
      v16h a = load_frag_lds(cs16 + (ti * 16 + (lane & 15)) * 64 + k0, lane);
      v16h b = load_frag_lds(q16 + (tj * 16 + (lane & 15)) * 64 + k0, lane);
      acc = wmma16(a, b, acc);
    }
    const int col = tj * 16 + (lane & 15);
    const int rbase = ti * 16 + ((lane >> 4) << 3);
    for (int r = 0; r < 8; ++r) {
      int row = rbase + r;
      S[row * 64 + col] = acc[r] + cw[row] + qw[col] + bcqv;
    }
  }
  __syncthreads();

  // phase 3: row softmax of S -> A16 (f16), row max -> smaxA
  if (tid < 64) {
    float mx = -1e30f;
    for (int j = 0; j < 64; ++j) mx = fmaxf(mx, S[tid * 64 + j]);
    smaxA[tid] = mx;
    float sum = 0.f;
    for (int j = 0; j < 64; ++j) { float e = expf(S[tid * 64 + j] - mx); sum += e; A16[tid * 64 + j] = (_Float16)e; }
    float inv = 1.f / sum;
    for (int j = 0; j < 64; ++j) A16[tid * 64 + j] = (_Float16)((float)A16[tid * 64 + j] * inv);
  }
  __syncthreads();

  // phase 4: b2 = softmax over rows of smax
  if (tid == 0) {
    float mx = -1e30f;
    for (int i = 0; i < 64; ++i) mx = fmaxf(mx, smaxA[i]);
    float sm = 0.f;
    for (int i = 0; i < 64; ++i) { float e = expf(smaxA[i] - mx); b2a[i] = e; sm += e; }
    float inv = 1.f / sm;
    for (int i = 0; i < 64; ++i) b2a[i] *= inv;
  }
  __syncthreads();

  // phase 5: c2q = A @ q (WMMA, overwrites S region);  q2c[h] = sum_i b2[i]*c[i][h]
  for (int t = 2 * wv; t < 2 * wv + 2; ++t) {
    const int ti = t >> 2, tj = t & 3;
    v8f acc = {};
    for (int k0 = 0; k0 < 64; k0 += 32) {
      v16h a = load_frag_lds(A16 + (ti * 16 + (lane & 15)) * 64 + k0, lane);
      v16h b = load_frag_lds(qT16 + (tj * 16 + (lane & 15)) * 64 + k0, lane);
      acc = wmma16(a, b, acc);
    }
    const int col = tj * 16 + (lane & 15);
    const int rbase = ti * 16 + ((lane >> 4) << 3);
    for (int r = 0; r < 8; ++r) c2q[(rbase + r) * 64 + col] = acc[r];
  }
  if (tid < 64) {
    float s = 0.f;
    for (int i = 0; i < 64; ++i) s += b2a[i] * cF[i * 64 + tid];
    q2c[tid] = s;
  }
  __syncthreads();

  // phase 6: sfeats partial dots + p1 = relu(g @ wz + bz)
  {
    const int k = tid >> 2, s = tid & 3;
    const float* sf = sfeats + ((long)m * KK + k) * FF;
    __builtin_prefetch(sf, 0, 1);
    float ps = 0.f;
    for (int f = s; f < FF; f += 4) ps += sf[f] * wloc[f];
    part[tid] = ps;
  }
  if (tid < 64) {
    float acc = bz[0];
    for (int h = 0; h < 64; ++h) {
      float cv = cF[tid * 64 + h], cq = c2q[tid * 64 + h];
      acc += cv * wz[h] + cq * wz[64 + h] + cv * cq * wz[128 + h] + cv * q2c[h] * wz[192 + h];
    }
    p1a[tid] = fmaxf(acc, 0.f);
  }
  __syncthreads();

  // phase 7: t = sfeats.wloc + p1*wloc[F] + bloc
  if (tid < 64)
    tv[tid] = bloc[0] + p1a[tid] * wloc[FF] +
              part[tid * 4] + part[tid * 4 + 1] + part[tid * 4 + 2] + part[tid * 4 + 3];
  __syncthreads();

  // phase 8: stats + the four normalizations (serial over 64, trivial cost)
  if (tid == 0) {
    float mu = 0.f;
    for (int i = 0; i < 64; ++i) mu += tv[i];
    mu *= (1.f / 64.f);
    float var = 0.f, tmin = 1e30f, tmax = -1e30f;
    for (int i = 0; i < 64; ++i) {
      float d = tv[i] - mu; var += d * d;
      tmin = fminf(tmin, tv[i]); tmax = fmaxf(tmax, tv[i]);
    }
    float sd = sqrtf(var / 63.f);                 // unbiased (ddof=1)
    float zmax = -1e30f;
    for (int i = 0; i < 64; ++i) { zA[i] = (tv[i] - mu) / sd; zmax = fmaxf(zmax, zA[i]); }
    float zs = 0.f;
    for (int i = 0; i < 64; ++i) { softA[i] = expf(zA[i] - zmax); zs += softA[i]; }
    for (int i = 0; i < 64; ++i) softA[i] /= zs;
    float smx = -1e30f;
    for (int i = 0; i < 64; ++i) smx = fmaxf(smx, softA[i]);
    float ss = 0.f;
    for (int i = 0; i < 64; ++i) { softsA[i] = expf(softA[i] - smx); ss += softsA[i]; }
    for (int i = 0; i < 64; ++i) softsA[i] /= ss;
    float us = 0.f;
    for (int i = 0; i < 64; ++i) { uniA[i] = (tv[i] + 1.f - tmin) / (tmax - tmin); us += uniA[i]; }
    for (int i = 0; i < 64; ++i) uniA[i] /= us;
  }
  __syncthreads();

  // phase 9: zero this mention's 4 output rows (d_out is poisoned by harness)
  for (int i = tid; i < 4 * NN; i += 256) {
    int s = i >> 11, col = i & (NN - 1);
    out[((long)s * MM + m) * NN + col] = 0.f;
  }
  __syncthreads();

  // phase 10: scatter-add (duplicates within the row need atomics)
  if (tid < 64) {
    int idx = cand_idx[m * KK + tid];
    atomicAdd(&out[((long)0 * MM + m) * NN + idx], zA[tid]);
    atomicAdd(&out[((long)1 * MM + m) * NN + idx], softA[tid]);
    atomicAdd(&out[((long)2 * MM + m) * NN + idx], uniA[tid]);
    atomicAdd(&out[((long)3 * MM + m) * NN + idx], softsA[tid]);
  }
}

// ---------------------------------------------------------------------------
// Host side
// ---------------------------------------------------------------------------
extern "C" void kernel_launch(void* const* d_in, const int* in_sizes, int n_in,
                              void* d_out, int out_size, void* d_ws, size_t ws_size,
                              hipStream_t stream) {
  const int*   title_vec   = (const int*)d_in[0];
  const int*   mention_vec = (const int*)d_in[1];
  const int*   cand_idx    = (const int*)d_in[2];
  const float* sfeats      = (const float*)d_in[3];
  const float* char_emb    = (const float*)d_in[4];
  const float* conv_w      = (const float*)d_in[5];
  const float* conv_b      = (const float*)d_in[6];
  const float* Wi_f        = (const float*)d_in[7];
  const float* Wh_f        = (const float*)d_in[8];
  const float* b_f         = (const float*)d_in[9];
  const float* Wi_b        = (const float*)d_in[10];
  const float* Wh_b        = (const float*)d_in[11];
  const float* b_b         = (const float*)d_in[12];
  const float* wc   = (const float*)d_in[13];
  const float* bc   = (const float*)d_in[14];
  const float* wq   = (const float*)d_in[15];
  const float* bq   = (const float*)d_in[16];
  const float* wcq  = (const float*)d_in[17];
  const float* bcq  = (const float*)d_in[18];
  const float* wz   = (const float*)d_in[19];
  const float* bz   = (const float*)d_in[20];
  const float* wloc = (const float*)d_in[21];
  const float* bloc = (const float*)d_in[22];

  // Workspace layout (~34.2 MB total), all offsets 256B-aligned
  char* ws = (char*)d_ws;
  _Float16* A16p  = (_Float16*)(ws);              // VP*EP*2      =  3,235,840
  _Float16* Bt    = (_Float16*)(ws +  3235840);   // 640*EP*2     =    409,600
  float*    P     = (float*)   (ws +  3645440);   // VP*640*4     = 12,943,360
  _Float16* T16   = (_Float16*)(ws + 16588800);   // 2048*128*2   =    524,288
  _Float16* Q16   = (_Float16*)(ws + 17113088);   // 256*128*2    =     65,536
  _Float16* WiTF  = (_Float16*)(ws + 17178624);   // 256*128*2    =     65,536
  _Float16* WhTF  = (_Float16*)(ws + 17244160);   // 256*64*2     =     32,768
  _Float16* WiTB  = (_Float16*)(ws + 17276928);   //                   65,536
  _Float16* WhTB  = (_Float16*)(ws + 17342464);   //                   32,768
  float*    hs    = (float*)   (ws + 17375232);   // 4*256*64*64*4= 16,777,216

  prep_A16<<<(VP * EP + 255) / 256, 256, 0, stream>>>(char_emb, A16p);
  prep_Bt<<<(640 * EP + 255) / 256, 256, 0, stream>>>(conv_w, Bt);
  transpose_f16<<<(128 * 256 + 255) / 256, 256, 0, stream>>>(Wi_f, WiTF, 128, 256);
  transpose_f16<<<(64 * 256 + 255) / 256, 256, 0, stream>>>(Wh_f, WhTF, 64, 256);
  transpose_f16<<<(128 * 256 + 255) / 256, 256, 0, stream>>>(Wi_b, WiTB, 128, 256);
  transpose_f16<<<(64 * 256 + 255) / 256, 256, 0, stream>>>(Wh_b, WhTB, 64, 256);

  tokproj_gemm<<<1580, 256, 0, stream>>>(A16p, Bt, P);

  conv_pool_kernel<<<NN, 128, 0, stream>>>(title_vec,   P, conv_b, T16, LT, LT - WW + 1);
  conv_pool_kernel<<<MM, 128, 0, stream>>>(mention_vec, P, conv_b, Q16, LM, LM - WW + 1);

  lstm_kernel<<<64, 256, 0, stream>>>(cand_idx, T16, Q16, WiTF, WhTF, WiTB, WhTB,
                                      b_f, b_b, hs);

  bidaf_kernel<<<MM, 256, 0, stream>>>(cand_idx, sfeats, hs,
                                       wc, bc, wq, bq, wcq, bcq, wz, bz, wloc, bloc,
                                       (float*)d_out);
}
